// MoCoModel_29291676958926
// MI455X (gfx1250) — matrix operand order repaired and checked by reading, not compiled, not run on hardware
//
#include <hip/hip_runtime.h>

// MoCo loss (InfoNCE + proto NCE) fused pipeline for gfx1250.
// Both large GEMMs are HBM-streaming-bound (queue=128MB, centroids=102MB @
// 23.3 TB/s); fp32 is converted to packed bf16 on the fly (v_add+v_perm_b32,
// 3 VALU ops per dword) and fed to v_wmma_f32_16x16x32_bf16 so compute stays
// ahead of the memory stream. B tiles are double-buffered through LDS (one
// barrier per K-chunk; next chunk's global loads issue before this chunk's
// WMMAs). Softmax over 65537 logits uses the analytic bound max <= 1/T
// (L2-normalized inputs) => single fused pass, no 64MB logits matrix.

#define C_DIM   512
#define B_ROWS  256
#define QUEUE_N 65536
#define NCL_N   50000
#define NNEG    512
#define INV_T   (1.0f / 0.07f)

typedef __attribute__((ext_vector_type(16))) __bf16        v16bf;
typedef __attribute__((ext_vector_type(2)))  __bf16        v2bf;
typedef __attribute__((ext_vector_type(8)))  float         v8f;
typedef __attribute__((ext_vector_type(8)))  unsigned int  v8u;

// two fp32 -> packed bf16 pair (one dword)
#if defined(__has_builtin) && __has_builtin(__builtin_amdgcn_cvt_pk_bf16_f32)
__device__ __forceinline__ unsigned int pack2bf(float x, float y) {
    v2bf r = __builtin_amdgcn_cvt_pk_bf16_f32(x, y);
    return __builtin_bit_cast(unsigned int, r);
}
#elif defined(__has_builtin) && __has_builtin(__builtin_amdgcn_perm)
// round-to-nearest via +0x8000, then v_perm_b32 packs the two high half-words
__device__ __forceinline__ unsigned int pack2bf(float x, float y) {
    const unsigned int ux = __float_as_uint(x) + 0x8000u;
    const unsigned int uy = __float_as_uint(y) + 0x8000u;
    return __builtin_amdgcn_perm(uy, ux, 0x07060302u);
}
#else
__device__ __forceinline__ unsigned int pack2bf(float x, float y) {
    unsigned int lo = __float_as_uint(x); lo += 0x7FFFu + ((lo >> 16) & 1u);
    unsigned int hi = __float_as_uint(y); hi += 0x7FFFu + ((hi >> 16) & 1u);
    return (lo >> 16) | (hi & 0xFFFF0000u);
}
#endif

// order-preserving float -> uint key (monotonic increasing)
__device__ __forceinline__ unsigned int fkey(float f) {
    unsigned int u = __float_as_uint(f);
    return ((int)u < 0) ? ~u : (u | 0x80000000u);
}

// ---------------------------------------------------------------------------
// K0: l_pos = rowdot(Z_q, Z_k); init rowSumExp with the positive term,
//     init argmin-packed, proto-loss rows, accuracy counter.
// ---------------------------------------------------------------------------
__global__ __launch_bounds__(256)
void init_kernel(const float* __restrict__ Zq, const float* __restrict__ Zk,
                 float* __restrict__ l_pos, float* __restrict__ rowSum,
                 unsigned long long* __restrict__ argminP,
                 unsigned int* __restrict__ colAcc, float* __restrict__ protoRow) {
    __shared__ float red[256];
    const int i = blockIdx.x, tid = threadIdx.x;
    const float* zq = Zq + (long long)i * C_DIM;
    const float* zk = Zk + (long long)i * C_DIM;
    red[tid] = zq[tid] * zk[tid] + zq[tid + 256] * zk[tid + 256];
    __syncthreads();
    for (int w = 128; w > 0; w >>= 1) {
        if (tid < w) red[tid] += red[tid + w];
        __syncthreads();
    }
    if (tid == 0) {
        float lp = red[0];
        l_pos[i]    = lp;
        rowSum[i]   = __expf(lp * INV_T - INV_T);   // positive-logit term, shifted
        argminP[i]  = ~0ull;
        protoRow[i] = 0.0f;
        if (i == 0) *colAcc = 0u;
    }
}

// ---------------------------------------------------------------------------
// K1/K2: fused bf16 WMMA GEMM  Z_q[256xC] @ Mat[ncols x C]^T.
//  Block: 8 waves, all 256 rows x 64 cols -> column reductions are local.
//  B chunks (64 cols x 32 K fp32) are cooperatively staged through a
//  double-buffered LDS tile with k contiguous per column (one barrier per chunk;
//  next chunk's global_load_b128s issue before this chunk's WMMAs).
//  Wave w: rows [32w,32w+32), 2x4 WMMA tiles; all 4 B fragments are hoisted
//  into registers before the WMMA chain so dscnt waits can be partial.
//  PROTO=false (queue): per-row sum exp(v/T - 1/T); per-col argmax accuracy.
//  PROTO=true (centroids): c_sq from the streamed fp32; per-row argmin of
//                          (c_sq - 2*dot) via packed u64 atomicMin.
// ---------------------------------------------------------------------------
template <bool PROTO>
__global__ __launch_bounds__(256)
void gemm_fused(const float* __restrict__ Zq, const float* __restrict__ Mat,
                int ncols,
                float* __restrict__ rowSum, unsigned int* __restrict__ colAcc,
                unsigned long long* __restrict__ argminP) {
    const int tid  = threadIdx.x;
    const int wave = tid >> 5;
    const int lane = tid & 31;
    const int lrow = lane & 15;
    const int hl   = lane >> 4;            // lane half (ISA bf16 fragment layout)
    const int n0   = blockIdx.x * 64;

    __shared__ unsigned int        ldsBpk[2][64 * 16];  // double-buffered B chunk
    __shared__ float               ldsRow[256];
    __shared__ unsigned int        ldsColKey[64];
    __shared__ float               ldsRow0[64];
    __shared__ float               ldsCsq[64];
    __shared__ unsigned long long  ldsMin[256];
    __shared__ unsigned int        ldsCnt;

    if (PROTO) {
        ldsMin[tid] = ~0ull;
        if (tid < 64) ldsCsq[tid] = 0.0f;
    } else {
        ldsRow[tid] = 0.0f;
        if (tid < 64) ldsColKey[tid] = 0u;
        if (tid == 0) ldsCnt = 0u;
    }

    const v8f vzero = {};
    v8f acc[2][4];
#pragma unroll
    for (int mt = 0; mt < 2; ++mt)
#pragma unroll
        for (int nt = 0; nt < 4; ++nt) acc[mt][nt] = vzero;

    // cooperative B-loader: thread owns float4 slots j = tid and tid+256;
    // slot j -> col (j>>3), K-quad (j&7)*4. Fixed cols => register csq.
    const int cbl0 = tid >> 3;              // col 0..31
    const int cbl1 = 32 + (tid >> 3);       // col 32..63
    const int kq   = (tid & 7) * 4;
    const int cg0  = n0 + cbl0, cg1 = n0 + cbl1;
    const long long bOff0 = (long long)((cg0 < ncols) ? cg0 : ncols - 1) * C_DIM + kq;
    const long long bOff1 = (long long)((cg1 < ncols) ? cg1 : ncols - 1) * C_DIM + kq;
    const int ldsIdx0 = cbl0 * 16 + (kq >> 1);
    const int ldsIdx1 = cbl1 * 16 + (kq >> 1);
    float csq0 = 0.f, csq1 = 0.f;

    const long long aOff0 = (long long)(wave * 32 + lrow) * C_DIM;
    const long long aOff1 = aOff0 + 16LL * C_DIM;

    // ---- prologue: stage chunk 0 into buffer 0 ----
    {
        const float4 f0 = *(const float4*)(Mat + bOff0);
        const float4 f1 = *(const float4*)(Mat + bOff1);
        if (PROTO) {
            csq0 += f0.x * f0.x + f0.y * f0.y + f0.z * f0.z + f0.w * f0.w;
            csq1 += f1.x * f1.x + f1.y * f1.y + f1.z * f1.z + f1.w * f1.w;
        }
        ldsBpk[0][ldsIdx0]     = pack2bf(f0.x, f0.y);
        ldsBpk[0][ldsIdx0 + 1] = pack2bf(f0.z, f0.w);
        ldsBpk[0][ldsIdx1]     = pack2bf(f1.x, f1.y);
        ldsBpk[0][ldsIdx1 + 1] = pack2bf(f1.z, f1.w);
    }
    __syncthreads();

#pragma unroll 1
    for (int kc = 0; kc < 16; ++kc) {
        const int  k0   = kc * 32;
        const int  p    = kc & 1;
        const bool more = (kc + 1) < 16;
        // ---- issue next chunk's global loads first (latency hides under WMMA)
        float4 nf0, nf1;
        if (more) {
            nf0 = *(const float4*)(Mat + bOff0 + k0 + 32);
            nf1 = *(const float4*)(Mat + bOff1 + k0 + 32);
            __builtin_prefetch(Mat + bOff0 + k0 + 64, 0, 1);
            __builtin_prefetch(Mat + bOff1 + k0 + 64, 0, 1);
        }
        // ---- A fragments (Z_q, L2-hot), ISA 16-bit A 16x32 layout ----
        // lane K-pairs: two contiguous 8-float runs at k0+8*hl and k0+16+8*hl
        v8u aU0, aU1;
        {
            const float* ap0 = Zq + aOff0 + k0 + 8 * hl;
            const float* ap1 = Zq + aOff1 + k0 + 8 * hl;
            float4 q0 = *(const float4*)(ap0);
            float4 q1 = *(const float4*)(ap0 + 4);
            float4 q2 = *(const float4*)(ap0 + 16);
            float4 q3 = *(const float4*)(ap0 + 20);
            aU0[0] = pack2bf(q0.x, q0.y); aU0[1] = pack2bf(q0.z, q0.w);
            aU0[2] = pack2bf(q1.x, q1.y); aU0[3] = pack2bf(q1.z, q1.w);
            aU0[4] = pack2bf(q2.x, q2.y); aU0[5] = pack2bf(q2.z, q2.w);
            aU0[6] = pack2bf(q3.x, q3.y); aU0[7] = pack2bf(q3.z, q3.w);
            q0 = *(const float4*)(ap1);
            q1 = *(const float4*)(ap1 + 4);
            q2 = *(const float4*)(ap1 + 16);
            q3 = *(const float4*)(ap1 + 20);
            aU1[0] = pack2bf(q0.x, q0.y); aU1[1] = pack2bf(q0.z, q0.w);
            aU1[2] = pack2bf(q1.x, q1.y); aU1[3] = pack2bf(q1.z, q1.w);
            aU1[4] = pack2bf(q2.x, q2.y); aU1[5] = pack2bf(q2.z, q2.w);
            aU1[6] = pack2bf(q3.x, q3.y); aU1[7] = pack2bf(q3.z, q3.w);
        }
        // ---- hoist all 4 B fragments from LDS buffer p ----
        v8u bU[4];
#pragma unroll
        for (int nt = 0; nt < 4; ++nt) {
            const uint4* bq = (const uint4*)&ldsBpk[p][(nt * 16 + lrow) * 16 + 4 * hl];
            const uint4 u0 = bq[0], u1 = bq[2];
            bU[nt][0] = u0.x; bU[nt][1] = u0.y; bU[nt][2] = u0.z; bU[nt][3] = u0.w;
            bU[nt][4] = u1.x; bU[nt][5] = u1.y; bU[nt][6] = u1.z; bU[nt][7] = u1.w;
        }
        const v16bf a0 = __builtin_bit_cast(v16bf, aU0);
        const v16bf a1 = __builtin_bit_cast(v16bf, aU1);
#pragma unroll
        for (int nt = 0; nt < 4; ++nt) {
            const v16bf b = __builtin_bit_cast(v16bf, bU[nt]);
            acc[0][nt] = __builtin_amdgcn_wmma_f32_16x16x32_bf16(
                false, a0, false, b, (short)0, acc[0][nt], false, false);
            acc[1][nt] = __builtin_amdgcn_wmma_f32_16x16x32_bf16(
                false, a1, false, b, (short)0, acc[1][nt], false, false);
        }
        // ---- stage next chunk into the other buffer ----
        if (more) {
            if (PROTO) {
                csq0 += nf0.x * nf0.x + nf0.y * nf0.y + nf0.z * nf0.z + nf0.w * nf0.w;
                csq1 += nf1.x * nf1.x + nf1.y * nf1.y + nf1.z * nf1.z + nf1.w * nf1.w;
            }
            unsigned int* q = ldsBpk[1 - p];
            q[ldsIdx0]     = pack2bf(nf0.x, nf0.y);
            q[ldsIdx0 + 1] = pack2bf(nf0.z, nf0.w);
            q[ldsIdx1]     = pack2bf(nf1.x, nf1.y);
            q[ldsIdx1 + 1] = pack2bf(nf1.z, nf1.w);
        }
        __syncthreads();   // stores visible for kc+1; reads of buf p done
    }

    // ---------------- epilogue ----------------
    if (PROTO) {
        atomicAdd(&ldsCsq[cbl0], csq0);   // 8 contributing threads per column
        atomicAdd(&ldsCsq[cbl1], csq1);
        __syncthreads();
#pragma unroll
        for (int mt = 0; mt < 2; ++mt)
#pragma unroll
            for (int nt = 0; nt < 4; ++nt) {
                const int cb = nt * 16 + lrow;
                const int c  = n0 + cb;
                if (c < ncols) {
                    const float cs = ldsCsq[cb];
#pragma unroll
                    for (int d = 0; d < 8; ++d) {
                        const int r = wave * 32 + mt * 16 + d + 8 * hl;
                        const float score = cs - 2.0f * acc[mt][nt][d];
                        const unsigned long long packed =
                            ((unsigned long long)fkey(score) << 32) | (unsigned int)c;
                        atomicMin(&ldsMin[r], packed);
                    }
                }
            }
        __syncthreads();
        atomicMin(&argminP[tid], ldsMin[tid]);
    } else {
#pragma unroll
        for (int mt = 0; mt < 2; ++mt)
#pragma unroll
            for (int nt = 0; nt < 4; ++nt) {
                const int cb = nt * 16 + lrow;
#pragma unroll
                for (int d = 0; d < 8; ++d) {
                    const int r = wave * 32 + mt * 16 + d + 8 * hl;
                    const float val = acc[mt][nt][d];
                    atomicAdd(&ldsRow[r], __expf(val * INV_T - INV_T));
                    if (r == 0) ldsRow0[cb] = val;                // single writer
                    else        atomicMax(&ldsColKey[cb], fkey(val));
                }
            }
        __syncthreads();
        atomicAdd(&rowSum[tid], ldsRow[tid]);
        if (tid < 64 && fkey(ldsRow0[tid]) >= ldsColKey[tid]) atomicAdd(&ldsCnt, 1u);
        __syncthreads();
        if (tid == 0) atomicAdd(colAcc, ldsCnt);
    }
}

// ---------------------------------------------------------------------------
// K3: proto NCE per row. Centroid rows are L2-resident (102MB < 192MB L2)
//     after the K2 stream, so the 256x512 gathered dots are L2-bound.
// ---------------------------------------------------------------------------
__global__ __launch_bounds__(256)
void proto_row(const float* __restrict__ Zq, const float* __restrict__ cent,
               const float* __restrict__ temps, const int* __restrict__ negraw,
               const unsigned long long* __restrict__ argminP,
               float* __restrict__ protoRow) {
    __shared__ float z[512];
    __shared__ float red[256];
    __shared__ float plposS, mxS;
    const int i = blockIdx.x, tid = threadIdx.x;
    const int I = (int)(argminP[i] & 0xFFFFFFFFull);

    const float* zq = Zq + (long long)i * C_DIM;
    z[tid] = zq[tid]; z[tid + 256] = zq[tid + 256];
    __syncthreads();

    // pl_pos = <z, centroid[I]> / temp[I]
    const float* cp = cent + (long long)I * C_DIM;
    red[tid] = z[tid] * cp[tid] + z[tid + 256] * cp[tid + 256];
    __syncthreads();
    for (int w = 128; w > 0; w >>= 1) { if (tid < w) red[tid] += red[tid + w]; __syncthreads(); }
    if (tid == 0) plposS = red[0] / temps[I];
    __syncthreads();

    // 2 negatives per thread (exclusion-shifted indices)
    float lg[2];
#pragma unroll
    for (int q = 0; q < 2; ++q) {
        const int k = tid + q * 256;
        int idx = negraw[(long long)i * NNEG + k];
        idx += (idx >= I) ? 1 : 0;
        const float4* c4 = (const float4*)(cent + (long long)idx * C_DIM);
        float dot = 0.f;
#pragma unroll 4
        for (int j = 0; j < 128; ++j) {
            const float4 c = c4[j];
            dot += z[4 * j] * c.x + z[4 * j + 1] * c.y + z[4 * j + 2] * c.z + z[4 * j + 3] * c.w;
        }
        lg[q] = dot / temps[idx];
    }
    // softmax over [pl_pos, 512 negs]
    red[tid] = fmaxf(lg[0], lg[1]); __syncthreads();
    for (int w = 128; w > 0; w >>= 1) { if (tid < w) red[tid] = fmaxf(red[tid], red[tid + w]); __syncthreads(); }
    if (tid == 0) mxS = fmaxf(red[0], plposS);
    __syncthreads();
    const float mx = mxS;
    red[tid] = __expf(lg[0] - mx) + __expf(lg[1] - mx); __syncthreads();
    for (int w = 128; w > 0; w >>= 1) { if (tid < w) red[tid] += red[tid + w]; __syncthreads(); }
    if (tid == 0) {
        const float sum = red[0] + __expf(plposS - mx);
        protoRow[i] = (mx + logf(sum)) - plposS;   // -log_softmax[0]
    }
}

// ---------------------------------------------------------------------------
// K4: combine. loss = mean(infoNCE) + 0.5*mean(proto); accuracy over 65537 cols.
// ---------------------------------------------------------------------------
__global__ __launch_bounds__(256)
void finalize(const float* __restrict__ l_pos, const float* __restrict__ rowSum,
              const unsigned int* __restrict__ colAcc,
              const float* __restrict__ protoRow, float* __restrict__ out) {
    __shared__ float red[256], red2[256];
    const int tid = threadIdx.x;
    const float li = (INV_T + logf(rowSum[tid])) - l_pos[tid] * INV_T;
    red[tid]  = li + 0.5f * protoRow[tid];
    red2[tid] = (tid == 0) ? -3.0e38f : l_pos[tid];   // max over rows >= 1
    __syncthreads();
    for (int w = 128; w > 0; w >>= 1) {
        if (tid < w) { red[tid] += red[tid + w]; red2[tid] = fmaxf(red2[tid], red2[tid + w]); }
        __syncthreads();
    }
    if (tid == 0) {
        out[0] = red[0] / 256.0f;
        const unsigned int cnt = *colAcc + ((l_pos[0] >= red2[0]) ? 1u : 0u);
        out[1] = (float)cnt / 65537.0f;
    }
}

// ---------------------------------------------------------------------------
extern "C" void kernel_launch(void* const* d_in, const int* in_sizes, int n_in,
                              void* d_out, int out_size, void* d_ws, size_t ws_size,
                              hipStream_t stream) {
    const float* Zq     = (const float*)d_in[0];
    const float* Zk     = (const float*)d_in[1];
    const float* queue  = (const float*)d_in[2];
    const float* cent   = (const float*)d_in[3];
    const float* temps  = (const float*)d_in[4];
    const int*   negraw = (const int*)d_in[5];
    float* out = (float*)d_out;

    char* ws = (char*)d_ws;
    float*              l_pos    = (float*)(ws + 0);
    float*              rowSum   = (float*)(ws + 1024);
    unsigned long long* argminP  = (unsigned long long*)(ws + 2048);
    float*              protoRow = (float*)(ws + 4096);
    unsigned int*       colAcc   = (unsigned int*)(ws + 5120);

    init_kernel<<<B_ROWS, 256, 0, stream>>>(Zq, Zk, l_pos, rowSum, argminP, colAcc, protoRow);
    gemm_fused<false><<<QUEUE_N / 64, 256, 0, stream>>>(Zq, queue, QUEUE_N, rowSum, colAcc, argminP);
    gemm_fused<true><<<(NCL_N + 63) / 64, 256, 0, stream>>>(Zq, cent, NCL_N, rowSum, colAcc, argminP);
    proto_row<<<B_ROWS, 256, 0, stream>>>(Zq, cent, temps, negraw, argminP, protoRow);
    finalize<<<1, 256, 0, stream>>>(l_pos, rowSum, colAcc, protoRow, out);
}
